// GraphEncoderGAT_71846212928191
// MI455X (gfx1250) — compile-verified, matrix operand order checked by reading, not run on hardware
//
#include <hip/hip_runtime.h>
#include <hip/hip_bf16.h>
#include <string.h>

typedef float  v2f   __attribute__((ext_vector_type(2)));
typedef float  v8f   __attribute__((ext_vector_type(8)));
typedef __bf16 v8bf  __attribute__((ext_vector_type(8)));
typedef __bf16 v16bf __attribute__((ext_vector_type(16)));

#define DEVINL __device__ __forceinline__

DEVINL unsigned enc_f(float f) {
    unsigned u = __float_as_uint(f);
    return (u >> 31) ? ~u : (u | 0x80000000u);
}
DEVINL float dec_f(unsigned o) {
    return (o >> 31) ? __uint_as_float(o & 0x7fffffffu) : __uint_as_float(~o);
}
DEVINL float leaky02(float a) { return a > 0.f ? a : 0.2f * a; }
DEVINL float elu1(float v)    { return v > 0.f ? v : (__expf(v) - 1.f); }

DEVINL void edge_sd(long long e, const int* __restrict__ src, const int* __restrict__ dst,
                    long long E, int& s, int& d) {
    if (e < E) { s = src[e]; d = dst[e]; }
    else       { s = d = (int)(e - E); }   // self loop
}

// ---------------- fill kernels (graph-capture-safe init) ----------------
__global__ void fill_f32(float* p, long long n, float v) {
    long long i = (long long)blockIdx.x * blockDim.x + threadIdx.x;
    if (i < n) p[i] = v;
}
__global__ void fill_u32(unsigned* p, long long n, unsigned v) {
    long long i = (long long)blockIdx.x * blockDim.x + threadIdx.x;
    if (i < n) p[i] = v;
}

// ---------------- fp32 WMMA GEMM: C[M,Nc] = A[M,K] @ B[K,Nc] (+bias, act) ----
// One wave per 16x64 output tile. grid = (M/16, Nc/64), block = 32.
// Uses V_WMMA_F32_16X16X4_F32 (exact fp32 accumulate).
__global__ void wmma_gemm_f32(const float* __restrict__ A, const float* __restrict__ B,
                              float* __restrict__ C, const float* __restrict__ bias,
                              int K, int Nc, int act) {
    const int lane = threadIdx.x;
    const int mt = blockIdx.x, nt = blockIdx.y;
    const int r0 = lane & 15;          // row within tile (A) / col within 16 (B,C)
    const int kh = (lane >> 4) << 1;   // 0 or 2 : K sub-offset per lane half
    const float* ap = A + (size_t)(mt * 16 + r0) * K + kh;

    v8f acc[4] = {};
    for (int k = 0; k < K; k += 4) {
        v2f a; a.x = ap[k]; a.y = ap[k + 1];
#pragma unroll
        for (int t = 0; t < 4; ++t) {
            const float* bp = B + (size_t)(k + kh) * Nc + nt * 64 + t * 16 + r0;
            v2f b; b.x = bp[0]; b.y = bp[Nc];
            acc[t] = __builtin_amdgcn_wmma_f32_16x16x4_f32(
                false, a, false, b, (short)0, acc[t], false, false);
        }
    }
    const int rbase = mt * 16 + ((lane >> 4) << 3);   // lanes 16-31 hold M=i+8
#pragma unroll
    for (int t = 0; t < 4; ++t) {
        const int col = nt * 64 + t * 16 + r0;
        const float bv = bias ? bias[col] : 0.f;
#pragma unroll
        for (int i = 0; i < 8; ++i) {
            float v = acc[t][i] + bv;
            if (act) v = fmaxf(v, 0.f);
            C[(size_t)(rbase + i) * Nc + col] = v;
        }
    }
}

// ---------------- bf16 WMMA GEMM with async global->LDS A staging ---------
// C[M,Nc] = A[M,K](bf16, row-major) @ Bt[Nc,K](bf16, pre-transposed).
// One wave per 16x64 output tile; the 16x32 A chunk (1KB) for each K-step is
// double-buffered through LDS with GLOBAL_LOAD_ASYNC_TO_LDS_B128 so HBM
// latency overlaps V_WMMA_F32_16X16X32_BF16 issue. grid=(M/16,Nc/64), block=32.
__global__ void wmma_gemm_bf16(const __bf16* __restrict__ A, const __bf16* __restrict__ Bt,
                               float* __restrict__ C, int K, int Nc) {
    __shared__ __attribute__((aligned(16))) char smem[2048];   // 2 x 1KB A buffers

    const int lane = threadIdx.x;
    const int mt = blockIdx.x, nt = blockIdx.y;
    const int r0 = lane & 15;
    const int kb = (lane >> 4) << 3;   // 0 or 8

    // async staging geometry: lane -> (row, 16B segment)
    const int rowL = lane >> 1;        // 0..15
    const int seg  = lane & 1;         // 0..1
    const unsigned ldsBase = (unsigned)(uintptr_t)smem;   // low 32 bits = LDS offset
    const __bf16* arow = A + (size_t)(mt * 16 + rowL) * K;

    auto issue_chunk = [&](int k, int buf) {
        unsigned l0 = ldsBase + (unsigned)(buf * 1024 + rowL * 64 + seg * 16);
        unsigned long long g0 =
            (unsigned long long)(uintptr_t)(arow + k + seg * 8);
        asm volatile("global_load_async_to_lds_b128 %0, %1, off"
                     :: "v"(l0), "v"(g0) : "memory");
        asm volatile("global_load_async_to_lds_b128 %0, %1, off"
                     :: "v"(l0 + 32u), "v"(g0 + 32ull) : "memory");
    };

    issue_chunk(0, 0);                          // prologue: stage chunk 0

    v8f acc[4] = {};
    int buf = 0;
    for (int k = 0; k < K; k += 32, buf ^= 1) {
        if (k + 32 < K) {
            // previous buffer's ds_loads must drain before async engine rewrites it
            asm volatile("s_wait_dscnt 0x0" ::: "memory");
            issue_chunk(k + 32, buf ^ 1);
            asm volatile("s_wait_asynccnt 0x2" ::: "memory");  // chunk k landed
        } else {
            asm volatile("s_wait_asynccnt 0x0" ::: "memory");  // last chunk landed
        }

        const char* lp = smem + buf * 1024 + r0 * 64 + kb * 2;
        v8bf alo = *(const v8bf*)lp;           // ds_load_b128
        v8bf ahi = *(const v8bf*)(lp + 32);    // ds_load_b128
        v16bf a = __builtin_shufflevector(alo, ahi,
                                          0,1,2,3,4,5,6,7,8,9,10,11,12,13,14,15);
#pragma unroll
        for (int t = 0; t < 4; ++t) {
            const __bf16* bp = Bt + (size_t)(nt * 64 + t * 16 + r0) * K + k + kb;
            v8bf blo = *(const v8bf*)bp;
            v8bf bhi = *(const v8bf*)(bp + 16);
            v16bf b = __builtin_shufflevector(blo, bhi,
                                              0,1,2,3,4,5,6,7,8,9,10,11,12,13,14,15);
            acc[t] = __builtin_amdgcn_wmma_f32_16x16x32_bf16(
                false, a, false, b, (short)0, acc[t], false, false);
        }
    }
    const int rbase = mt * 16 + ((lane >> 4) << 3);
#pragma unroll
    for (int t = 0; t < 4; ++t) {
        const int col = nt * 64 + t * 16 + r0;
#pragma unroll
        for (int i = 0; i < 8; ++i)
            C[(size_t)(rbase + i) * Nc + col] = acc[t][i];
    }
}

// ---------------- attention per-node dot products -------------------------
// as[n,h] = <h[n,h,:], att_src[h,:]> ; ad likewise. C = 64.
__global__ void att_dots(const float* __restrict__ h, const float* __restrict__ wsrc,
                         const float* __restrict__ wdst, float* __restrict__ osrc,
                         float* __restrict__ odst, long long NH, int H) {
    long long i = (long long)blockIdx.x * blockDim.x + threadIdx.x;
    if (i >= NH) return;
    const int hh = (int)(i % H);
    const float* hp = h + i * 64;              // [N,H,64] flattened == i*64
    const float* s = wsrc + hh * 64;
    const float* d = wdst + hh * 64;
    float vs = 0.f, vd = 0.f;
#pragma unroll 8
    for (int c = 0; c < 64; ++c) { float x = hp[c]; vs += x * s[c]; vd += x * d[c]; }
    osrc[i] = vs; odst[i] = vd;
}

// ---------------- edge softmax pass 1: per-dst running max ----------------
__global__ void edge_max(const int* __restrict__ src, const int* __restrict__ dst,
                         long long E, long long tot,
                         const float* __restrict__ as, const float* __restrict__ ad,
                         unsigned* __restrict__ menc, int H) {
    long long i = (long long)blockIdx.x * blockDim.x + threadIdx.x;
    if (i >= tot) return;
    long long e = i / H; int hh = (int)(i % H);
    int s, d; edge_sd(e, src, dst, E, s, d);
    float a = leaky02(as[(size_t)s * H + hh] + ad[(size_t)d * H + hh]);
    atomicMax(&menc[(size_t)d * H + hh], enc_f(a));
}

// ---------------- edge softmax pass 2: exp + per-dst sum ------------------
__global__ void edge_exp(const int* __restrict__ src, const int* __restrict__ dst,
                         long long E, long long tot,
                         const float* __restrict__ as, const float* __restrict__ ad,
                         const unsigned* __restrict__ menc,
                         float* __restrict__ ebuf, float* __restrict__ ssum, int H) {
    long long i = (long long)blockIdx.x * blockDim.x + threadIdx.x;
    if (i >= tot) return;
    long long e = i / H; int hh = (int)(i % H);
    int s, d; edge_sd(e, src, dst, E, s, d);
    float a = leaky02(as[(size_t)s * H + hh] + ad[(size_t)d * H + hh]);
    float m = dec_f(menc[(size_t)d * H + hh]);
    float ex = __expf(a - m);
    ebuf[i] = ex;
    atomicAdd(&ssum[(size_t)d * H + hh], ex);
}

// ---------------- edge softmax pass 3: weighted scatter-add ---------------
// thread per (edge, feature); F = H*64, head = feature>>6.
__global__ void edge_agg(const int* __restrict__ src, const int* __restrict__ dst,
                         long long E, long long tot,
                         const float* __restrict__ h, const float* __restrict__ ebuf,
                         const float* __restrict__ ssum, float* __restrict__ out, int H) {
    long long i = (long long)blockIdx.x * blockDim.x + threadIdx.x;
    const long long F = (long long)H * 64;
    if (i >= tot) return;
    long long e = i / F; int j = (int)(i % F); int hh = j >> 6;
    int s, d; edge_sd(e, src, dst, E, s, d);
    float w = ebuf[e * H + hh] / ssum[(size_t)d * H + hh];
    atomicAdd(&out[(size_t)d * F + j], h[(size_t)s * F + j] * w);
}

// ---------------- layer-1 epilogue: +b1, ELU, cast to bf16 ----------------
__global__ void epi1_bf16(const float* __restrict__ out1, const float* __restrict__ b1,
                          __bf16* __restrict__ x2b, long long n) {
    long long i = (long long)blockIdx.x * blockDim.x + threadIdx.x;
    if (i >= n) return;
    x2b[i] = (__bf16)elu1(out1[i] + b1[i % 640]);
}

// ---------------- W2 -> bf16 transpose ([640,64] -> [64,640]) -------------
__global__ void conv_w2t(const float* __restrict__ W2, __bf16* __restrict__ W2t,
                         int K, int Nc) {
    int i = blockIdx.x * blockDim.x + threadIdx.x;
    if (i >= K * Nc) return;
    int k = i / Nc, c = i % Nc;
    W2t[(size_t)c * K + k] = (__bf16)W2[i];
}

// ---------------- layer-2 epilogue fused with pooling sum -----------------
__global__ void epi2_pool(const float* __restrict__ out2, const float* __restrict__ b2,
                          const int* __restrict__ batch, float* __restrict__ sums,
                          long long n) {
    long long i = (long long)blockIdx.x * blockDim.x + threadIdx.x;
    if (i >= n) return;
    int c = (int)(i & 63); int node = (int)(i >> 6);
    float v = elu1(out2[i] + b2[c]);
    atomicAdd(&sums[(size_t)batch[node] * 64 + c], v);
}
__global__ void count_nodes(const int* __restrict__ batch, float* __restrict__ cnts, int N) {
    int i = blockIdx.x * blockDim.x + threadIdx.x;
    if (i < N) atomicAdd(&cnts[batch[i]], 1.0f);
}
__global__ void pool_div(const float* __restrict__ sums, const float* __restrict__ cnts,
                         float* __restrict__ pooled, int n) {
    int i = blockIdx.x * blockDim.x + threadIdx.x;
    if (i < n) pooled[i] = sums[i] / fmaxf(cnts[i >> 6], 1.0f);
}

// =========================================================================
extern "C" void kernel_launch(void* const* d_in, const int* in_sizes, int n_in,
                              void* d_out, int out_size, void* d_ws, size_t ws_size,
                              hipStream_t stream) {
    const float* x    = (const float*)d_in[0];
    const int*   ei   = (const int*)  d_in[1];
    const int*   batch= (const int*)  d_in[2];
    const float* W1   = (const float*)d_in[3];
    const float* As1  = (const float*)d_in[4];
    const float* Ad1  = (const float*)d_in[5];
    const float* b1   = (const float*)d_in[6];
    const float* W2   = (const float*)d_in[7];
    const float* As2  = (const float*)d_in[8];
    const float* Ad2  = (const float*)d_in[9];
    const float* b2   = (const float*)d_in[10];
    const float* Wh1  = (const float*)d_in[11];
    const float* bh1  = (const float*)d_in[12];
    const float* Wh2  = (const float*)d_in[13];
    const float* bh2  = (const float*)d_in[14];

    const int       N  = in_sizes[0] / 32;      // 100000 (multiple of 16)
    const long long E  = in_sizes[1] / 2;       // 400000
    const long long ET = E + N;                 // + self loops
    const int       G  = out_size / 768;        // 2000 (multiple of 16)
    const int H1 = 10, F1 = 640, F2 = 64;

    const int* srcp = ei;
    const int* dstp = ei + E;

    // ---- workspace carve-out (256B aligned) ----
    char* ws = (char*)d_ws;
    size_t off = 0;
    auto carve = [&](size_t bytes) -> void* {
        void* p = ws + off;
        off = (off + bytes + 255) & ~(size_t)255;
        return p;
    };
    float*    h1   = (float*)   carve((size_t)N * F1 * 4);
    float*    out1 = (float*)   carve((size_t)N * F1 * 4);
    __bf16*   x2b  = (__bf16*)  carve((size_t)N * F1 * 2);
    float*    as1  = (float*)   carve((size_t)N * H1 * 4);
    float*    ad1  = (float*)   carve((size_t)N * H1 * 4);
    unsigned* m1   = (unsigned*)carve((size_t)N * H1 * 4);
    float*    s1   = (float*)   carve((size_t)N * H1 * 4);
    float*    eb1  = (float*)   carve((size_t)ET * H1 * 4);
    float*    h2   = (float*)   carve((size_t)N * F2 * 4);
    float*    out2 = (float*)   carve((size_t)N * F2 * 4);
    float*    as2  = (float*)   carve((size_t)N * 4);
    float*    ad2  = (float*)   carve((size_t)N * 4);
    unsigned* m2   = (unsigned*)carve((size_t)N * 4);
    float*    s2   = (float*)   carve((size_t)N * 4);
    float*    eb2  = (float*)   carve((size_t)ET * 4);
    __bf16*   W2t  = (__bf16*)  carve((size_t)F1 * F2 * 2);
    float*    sums = (float*)   carve((size_t)G * 64 * 4);
    float*    cnts = (float*)   carve((size_t)G * 4);
    float*    pooled=(float*)   carve((size_t)G * 64 * 4);
    float*    hidden=(float*)   carve((size_t)G * 256 * 4);
    (void)ws_size; (void)n_in;

    // encoded(-3e38) for the ordered-uint float-max trick
    float ninf = -3.0e38f; unsigned uraw; memcpy(&uraw, &ninf, 4);
    const unsigned mInit = ~uraw;   // sign bit set -> enc = ~bits

    auto blocks = [](long long n) { return (unsigned)((n + 255) / 256); };

    // ---- init accumulators ----
    fill_f32<<<blocks((long long)N * F1), 256, 0, stream>>>(out1, (long long)N * F1, 0.f);
    fill_f32<<<blocks((long long)N * H1), 256, 0, stream>>>(s1, (long long)N * H1, 0.f);
    fill_u32<<<blocks((long long)N * H1), 256, 0, stream>>>(m1, (long long)N * H1, mInit);
    fill_f32<<<blocks((long long)N * F2), 256, 0, stream>>>(out2, (long long)N * F2, 0.f);
    fill_f32<<<blocks(N), 256, 0, stream>>>(s2, N, 0.f);
    fill_u32<<<blocks(N), 256, 0, stream>>>(m2, N, mInit);
    fill_f32<<<blocks((long long)G * 64), 256, 0, stream>>>(sums, (long long)G * 64, 0.f);
    fill_f32<<<blocks(G), 256, 0, stream>>>(cnts, G, 0.f);
    conv_w2t<<<blocks(F1 * F2), 256, 0, stream>>>(W2, W2t, F1, F2);

    // ---- layer 1: h1 = x @ W1 (fp32 WMMA, K=32) ----
    wmma_gemm_f32<<<dim3(N / 16, F1 / 64), 32, 0, stream>>>(x, W1, h1, nullptr, 32, F1, 0);
    att_dots<<<blocks((long long)N * H1), 256, 0, stream>>>(h1, As1, Ad1, as1, ad1,
                                                            (long long)N * H1, H1);
    edge_max<<<blocks(ET * H1), 256, 0, stream>>>(srcp, dstp, E, ET * H1, as1, ad1, m1, H1);
    edge_exp<<<blocks(ET * H1), 256, 0, stream>>>(srcp, dstp, E, ET * H1, as1, ad1, m1,
                                                  eb1, s1, H1);
    edge_agg<<<blocks(ET * F1), 256, 0, stream>>>(srcp, dstp, E, ET * F1, h1, eb1, s1,
                                                  out1, H1);
    epi1_bf16<<<blocks((long long)N * F1), 256, 0, stream>>>(out1, b1, x2b, (long long)N * F1);

    // ---- layer 2: h2 = elu(out1+b1) @ W2 (bf16 WMMA + async-LDS staging) ----
    wmma_gemm_bf16<<<dim3(N / 16, F2 / 64), 32, 0, stream>>>(x2b, W2t, h2, F1, F2);
    att_dots<<<blocks(N), 256, 0, stream>>>(h2, As2, Ad2, as2, ad2, N, 1);
    edge_max<<<blocks(ET), 256, 0, stream>>>(srcp, dstp, E, ET, as2, ad2, m2, 1);
    edge_exp<<<blocks(ET), 256, 0, stream>>>(srcp, dstp, E, ET, as2, ad2, m2, eb2, s2, 1);
    edge_agg<<<blocks(ET * F2), 256, 0, stream>>>(srcp, dstp, E, ET * F2, h2, eb2, s2,
                                                  out2, 1);

    // ---- pooling + MLP head (fp32 WMMA) ----
    epi2_pool<<<blocks((long long)N * F2), 256, 0, stream>>>(out2, b2, batch, sums,
                                                             (long long)N * F2);
    count_nodes<<<blocks(N), 256, 0, stream>>>(batch, cnts, N);
    pool_div<<<blocks((long long)G * 64), 256, 0, stream>>>(sums, cnts, pooled, G * 64);

    wmma_gemm_f32<<<dim3(G / 16, 256 / 64), 32, 0, stream>>>(pooled, Wh1, hidden, bh1,
                                                             64, 256, 1);
    wmma_gemm_f32<<<dim3(G / 16, 768 / 64), 32, 0, stream>>>(hidden, Wh2, (float*)d_out,
                                                             bh2, 256, 768, 0);
}
